// RelPositionMultiHeadAttention_8804682957429
// MI455X (gfx1250) — matrix operand order repaired and checked by reading, not compiled, not run on hardware
//
#include <hip/hip_runtime.h>

// Problem constants (from reference): B=4, T=K=2048, E=1024, H=16, D=64
#define BB   4
#define TT   2048
#define EE   1024
#define HH   16
#define DD   64
#define HDIM 1024      // H*D
#define M1   8192      // T*B
#define SSS  2047      // T-1

typedef __attribute__((ext_vector_type(16))) __bf16 v16bf;
typedef __attribute__((ext_vector_type(8)))  __bf16 v8bf;
typedef __attribute__((ext_vector_type(8)))  float  v8f;

__device__ __forceinline__ v8f wmma_bf16(v16bf a, v16bf b, v8f c) {
  return __builtin_amdgcn_wmma_f32_16x16x32_bf16(false, a, false, b, (short)0, c,
                                                 false, false);
}

__device__ __forceinline__ v16bf cat16(v8bf lo, v8bf hi) {
  return __builtin_shufflevector(lo, hi, 0, 1, 2, 3, 4, 5, 6, 7, 8, 9, 10, 11, 12,
                                 13, 14, 15);
}

// A operand 16x32 bf16: lane<16 holds row l, K={0..15};
// lane>=16 row l-16, K={16..31}.  Two 16B chunks per lane.
__device__ __forceinline__ v16bf loadA16x32(const __bf16* base, int ld, int row0,
                                            int k0) {
  const int l = threadIdx.x & 31;
  const int off = (l >> 4) * 16;
  const __bf16* p = base + (long)(row0 + (l & 15)) * ld + k0 + off;
  v8bf lo = *(const v8bf*)p;
  v8bf hi = *(const v8bf*)(p + 8);
  return cat16(lo, hi);
}

// B operand 32x16 bf16 from row-major B^T storage [N, K]: lane holds column
// (l&15) with 16 contiguous K values at chunk (l>>4)*16.  One 32B load.
__device__ __forceinline__ v16bf loadB32x16(const __bf16* base, int ld, int col0,
                                            int k0) {
  const int l = threadIdx.x & 31;
  const __bf16* p = base + (long)(col0 + (l & 15)) * ld + k0 + (l >> 4) * 16;
  return *(const v16bf*)p;
}

// ---------------- conversion / setup kernels ----------------

__global__ void cvt_f32_bf16(const float* __restrict__ in, __bf16* __restrict__ out,
                             long n) {
  long i = (long)blockIdx.x * blockDim.x + threadIdx.x;
  if (i < n) out[i] = (__bf16)in[i];
}

// word_embedding [B,T,E] f32 -> we_bf16 [(t*B+b), E]
__global__ void cvt_we(const float* __restrict__ we, __bf16* __restrict__ out) {
  long idx = (long)blockIdx.x * blockDim.x + threadIdx.x;
  if (idx >= (long)M1 * EE) return;
  int e = (int)(idx & (EE - 1));
  long m = idx >> 10;
  int b = (int)(m & (BB - 1));
  long t = m >> 2;
  out[idx] = (__bf16)we[((long)b * TT + t) * EE + e];
}

// sinusoidal positional encoding table, bf16 [T, E]
__global__ void pe_build(__bf16* __restrict__ pe) {
  long idx = (long)blockIdx.x * blockDim.x + threadIdx.x;
  if (idx >= (long)TT * EE) return;
  int e = (int)(idx & (EE - 1));
  long k = idx >> 10;
  float div = __expf((float)(e & ~1) * (-9.210340371976184f / (float)EE));
  float v = (float)k * div;
  pe[idx] = (__bf16)((e & 1) ? cosf(v) : sinf(v));
}

// ---------------- NT bf16 WMMA GEMM:  C[M,N] = A[M,K] * B[N,K]^T ----------------
// block = 8 waves, wave tile 32x32 (4 accumulators), block tile 128x64.
// mode 0: store bf16 row-major (ldc).  mode 1: scatter f32 to d_out [B,T,E]
// from row m = t*B+b (output projection, fused transpose).
__global__ __launch_bounds__(256) void gemm_nt_kernel(
    const __bf16* __restrict__ A, int lda, const __bf16* __restrict__ Bt, int ldb,
    int Kd, __bf16* __restrict__ Cb, int ldc, float* __restrict__ Cf, int mode) {
  const int w = threadIdx.x >> 5;
  const int l = threadIdx.x & 31;
  const int hi = l >> 4, ln = l & 15;
  const int m0 = blockIdx.x * 128 + (w & 3) * 32;
  const int n0 = blockIdx.y * 64 + (w >> 2) * 32;

  v8f c00 = {}, c01 = {}, c10 = {}, c11 = {};
  for (int k = 0; k < Kd; k += 32) {
    v16bf a0 = loadA16x32(A, lda, m0, k);
    v16bf a1 = loadA16x32(A, lda, m0 + 16, k);
    v16bf b0 = loadB32x16(Bt, ldb, n0, k);
    v16bf b1 = loadB32x16(Bt, ldb, n0 + 16, k);
    c00 = wmma_bf16(a0, b0, c00);
    c01 = wmma_bf16(a0, b1, c01);
    c10 = wmma_bf16(a1, b0, c10);
    c11 = wmma_bf16(a1, b1, c11);
  }

#pragma unroll
  for (int ti = 0; ti < 4; ++ti) {
    v8f cc = (ti == 0) ? c00 : (ti == 1) ? c01 : (ti == 2) ? c10 : c11;
    int rb = m0 + (ti >> 1) * 16;
    int cb = n0 + (ti & 1) * 16;
#pragma unroll
    for (int g = 0; g < 8; ++g) {
      int row = rb + g + hi * 8;  // C layout: VGPR g holds M=g / M=g+8
      int col = cb + ln;
      float v = cc[g];
      if (mode == 0) {
        Cb[(long)row * ldc + col] = (__bf16)v;
      } else {
        int bq = row & (BB - 1);
        long tq = row >> 2;
        Cf[((long)bq * TT + tq) * EE + col] = v;
      }
    }
  }
}

// ---------------- prep: split heads, add biases, relayout ----------------
// heads bf16 [m=t*B+b, 3*HD] ->
//   qw/qr/kk [bh, t, d] bf16 ; vt [bh, d, t] bf16 (transposed for PV B-operand)
//   Rkh [h, t, d] from rk_tmp [t, HD]
__global__ void prep_kernel(const __bf16* __restrict__ heads,
                            const __bf16* __restrict__ rk_tmp,
                            const float* __restrict__ r_w,
                            const float* __restrict__ r_r, __bf16* __restrict__ qw,
                            __bf16* __restrict__ qr, __bf16* __restrict__ kk,
                            __bf16* __restrict__ vt, __bf16* __restrict__ Rkh) {
  long idx = (long)blockIdx.x * blockDim.x + threadIdx.x;
  if (idx >= (long)BB * HH * TT * DD) return;
  int d = (int)(idx & (DD - 1));
  int t = (int)((idx >> 6) & (TT - 1));
  int bh = (int)(idx >> 17);
  int b = bh >> 4, h = bh & (HH - 1);
  long mrow = (long)t * BB + b;
  const __bf16* hb = heads + mrow * (3 * HDIM);
  int f = h * DD + d;
  float qv = (float)hb[f];
  qw[idx] = (__bf16)(qv + r_w[f]);
  qr[idx] = (__bf16)(qv + r_r[f]);
  kk[idx] = hb[HDIM + f];
  vt[((long)bh * DD + d) * TT + t] = hb[2 * HDIM + f];
  if (b == 0) Rkh[((long)h * TT + t) * DD + d] = rk_tmp[(long)t * HDIM + f];
}

// ---------------- fused flash attention with rel-shift ----------------
// grid (T/128, B*H); 8 waves/block, each wave owns a 16-row query tile and
// streams 32 keys/iter.  BD via psi[i,m]=qr[i]*Rk'[m] band (3-tile LDS ring,
// 2 fresh WMMA tiles per iter); AC via WMMA; online softmax; PV via WMMA.
__global__ __launch_bounds__(256) void attn_kernel(
    const __bf16* __restrict__ qw, const __bf16* __restrict__ qr,
    const __bf16* __restrict__ kk, const __bf16* __restrict__ vt,
    const __bf16* __restrict__ Rkh, const int* __restrict__ lens,
    __bf16* __restrict__ vec) {
  __shared__ float strip[8][16][48];   // per-wave psi ring: 16 rows x 3 tiles
  __shared__ __bf16 pbuf[8][16][32];   // per-wave P relayout buffer

  const int w = threadIdx.x >> 5;
  const int l = threadIdx.x & 31;
  const int hi = l >> 4, ln = l & 15;
  const int bh = blockIdx.y;
  const int b = bh >> 4, h = bh & (HH - 1);
  const int i0 = blockIdx.x * 128 + w * 16;
  const int len = lens[b];

  const __bf16* qwB = qw + (long)bh * TT * DD;
  const __bf16* qrB = qr + (long)bh * TT * DD;
  const __bf16* kkB = kk + (long)bh * TT * DD;
  const __bf16* vtB = vt + (long)bh * DD * TT;
  const __bf16* RkB = Rkh + (long)h * TT * DD;

  const v16bf qwA0 = loadA16x32(qwB, DD, i0, 0);
  const v16bf qwA1 = loadA16x32(qwB, DD, i0, 32);
  const v16bf qrA0 = loadA16x32(qrB, DD, i0, 0);
  const v16bf qrA1 = loadA16x32(qrB, DD, i0, 32);

  v8f acc0 = {}, acc1 = {}, acc2 = {}, acc3 = {};
  float mrow[8], lrow[8];
#pragma unroll
  for (int g = 0; g < 8; ++g) { mrow[g] = -3.0e38f; lrow[g] = 0.0f; }

  const v16bf zv = {};

  // one 16x16 psi tile at (16-aligned) extended-index base mt; Rk'[m] gather
  // folds the TXL rel-shift wrap (m==T -> zero column) into the B operand.
  auto phi_tile = [&](int mt) {
    int mc = mt + ln;
    int kx = (mc < TT) ? mc : (mc - TT - 1);
    kx = (kx < 0) ? 0 : ((kx > TT - 1) ? TT - 1 : kx);
    const __bf16* rp = RkB + (long)kx * DD + hi * 16;
    v16bf rb0 = *(const v16bf*)(rp);
    v16bf rb1 = *(const v16bf*)(rp + 32);
    if (mc == TT) { rb0 = zv; rb1 = zv; }
    v8f pc = {};
    pc = wmma_bf16(qrA0, rb0, pc);
    pc = wmma_bf16(qrA1, rb1, pc);
    const int col = ((mt >> 4) % 3) * 16 + ln;
#pragma unroll
    for (int g = 0; g < 8; ++g) strip[w][g + hi * 8][col] = pc[g];
  };

  const float scale = 0.125f;  // 1/sqrt(64)

  for (int j0 = 0; j0 < TT; j0 += 32) {
    if (j0 >= len) break;  // fully masked tail
    const int bm = SSS - i0 - 15 + j0;  // 16-aligned band base
    if (j0 == 0) phi_tile(bm);          // ring warm-up
    phi_tile(bm + 16);
    phi_tile(bm + 32);

    // AC = (q + r_w_bias) K^T for 2 key sub-tiles
    v8f s0 = {}, s1 = {};
    {
      v16bf kb0 = loadB32x16(kkB, DD, j0, 0);
      v16bf kb1 = loadB32x16(kkB, DD, j0, 32);
      s0 = wmma_bf16(qwA0, kb0, s0);
      s0 = wmma_bf16(qwA1, kb1, s0);
      v16bf kc0 = loadB32x16(kkB, DD, j0 + 16, 0);
      v16bf kc1 = loadB32x16(kkB, DD, j0 + 16, 32);
      s1 = wmma_bf16(qwA0, kc0, s1);
      s1 = wmma_bf16(qwA1, kc1, s1);
    }

#pragma unroll
    for (int g = 0; g < 8; ++g) {
      const int r = g + hi * 8;
      int ma = bm + 15 - r + ln;  // m index for (row r, col j0+ln)
      int mb2 = ma + 16;
      float bd0 = strip[w][r][((ma >> 4) % 3) * 16 + (ma & 15)];
      float bd1 = strip[w][r][((mb2 >> 4) % 3) * 16 + (mb2 & 15)];
      float v0 = (s0[g] + bd0) * scale;
      float v1 = (s1[g] + bd1) * scale;
      if (j0 + ln >= len) v0 = -1.0e30f;
      if (j0 + 16 + ln >= len) v1 = -1.0e30f;

      float mx = fmaxf(v0, v1);
#pragma unroll
      for (int d = 8; d >= 1; d >>= 1) mx = fmaxf(mx, __shfl_xor(mx, d, 32));
      float mnew = fmaxf(mrow[g], mx);
      float corr = __expf(mrow[g] - mnew);
      float p0 = __expf(v0 - mnew);
      float p1 = __expf(v1 - mnew);
      float rs = p0 + p1;
#pragma unroll
      for (int d = 8; d >= 1; d >>= 1) rs += __shfl_xor(rs, d, 32);
      lrow[g] = lrow[g] * corr + rs;
      mrow[g] = mnew;
      acc0[g] *= corr; acc1[g] *= corr; acc2[g] *= corr; acc3[g] *= corr;
      pbuf[w][r][ln] = (__bf16)p0;
      pbuf[w][r][16 + ln] = (__bf16)p1;
    }

    // reload P in A-operand layout (wave-private LDS, DS ops are in order)
    {
      const int off = hi * 16;
      v8bf plo = *(const v8bf*)&pbuf[w][ln][off];
      v8bf phh = *(const v8bf*)&pbuf[w][ln][8 + off];
      v16bf pa = cat16(plo, phh);
      v16bf vb0 = loadB32x16(vtB, TT, 0, j0);
      v16bf vb1 = loadB32x16(vtB, TT, 16, j0);
      v16bf vb2 = loadB32x16(vtB, TT, 32, j0);
      v16bf vb3 = loadB32x16(vtB, TT, 48, j0);
      acc0 = wmma_bf16(pa, vb0, acc0);
      acc1 = wmma_bf16(pa, vb1, acc1);
      acc2 = wmma_bf16(pa, vb2, acc2);
      acc3 = wmma_bf16(pa, vb3, acc3);
    }
  }

#pragma unroll
  for (int g = 0; g < 8; ++g) {
    const float inv = 1.0f / lrow[g];
    const int t = i0 + g + hi * 8;
    __bf16* o = vec + ((long)t * BB + b) * HDIM + h * DD;
    o[ln] = (__bf16)(acc0[g] * inv);
    o[16 + ln] = (__bf16)(acc1[g] * inv);
    o[32 + ln] = (__bf16)(acc2[g] * inv);
    o[48 + ln] = (__bf16)(acc3[g] * inv);
  }
}

// ---------------- host-side orchestration ----------------

extern "C" void kernel_launch(void* const* d_in, const int* in_sizes, int n_in,
                              void* d_out, int out_size, void* d_ws, size_t ws_size,
                              hipStream_t stream) {
  (void)in_sizes; (void)n_in; (void)out_size; (void)ws_size;
  const float* we = (const float*)d_in[0];
  const int* lens = (const int*)d_in[1];
  const float* Wqkv = (const float*)d_in[2];
  const float* Wpos = (const float*)d_in[3];
  const float* Wout = (const float*)d_in[4];
  const float* r_w = (const float*)d_in[5];
  const float* r_r = (const float*)d_in[6];
  float* out = (float*)d_out;

  char* ws = (char*)d_ws;
  size_t off = 0;
  auto alloc = [&](size_t bytes) -> void* {
    void* p = ws + off;
    off += (bytes + 255) & ~(size_t)255;
    return p;
  };
  __bf16* we_b   = (__bf16*)alloc((size_t)M1 * EE * 2);
  __bf16* wqkv_b = (__bf16*)alloc((size_t)3 * HDIM * EE * 2);
  __bf16* wpos_b = (__bf16*)alloc((size_t)HDIM * EE * 2);
  __bf16* wout_b = (__bf16*)alloc((size_t)EE * HDIM * 2);
  __bf16* pe_b   = (__bf16*)alloc((size_t)TT * EE * 2);
  __bf16* heads_b = (__bf16*)alloc((size_t)M1 * 3 * HDIM * 2);
  __bf16* rk_b   = (__bf16*)alloc((size_t)TT * HDIM * 2);
  __bf16* qw_b   = (__bf16*)alloc((size_t)BB * HH * TT * DD * 2);
  __bf16* qr_b   = (__bf16*)alloc((size_t)BB * HH * TT * DD * 2);
  __bf16* kk_b   = (__bf16*)alloc((size_t)BB * HH * TT * DD * 2);
  __bf16* vt_b   = (__bf16*)alloc((size_t)BB * HH * DD * TT * 2);
  __bf16* rkh_b  = (__bf16*)alloc((size_t)HH * TT * DD * 2);
  __bf16* vec_b  = (__bf16*)alloc((size_t)M1 * HDIM * 2);

  const long nWE = (long)M1 * EE;
  cvt_we<<<(unsigned)((nWE + 255) / 256), 256, 0, stream>>>(we, we_b);
  cvt_f32_bf16<<<(unsigned)(((long)3 * HDIM * EE + 255) / 256), 256, 0, stream>>>(
      Wqkv, wqkv_b, (long)3 * HDIM * EE);
  cvt_f32_bf16<<<(unsigned)(((long)HDIM * EE + 255) / 256), 256, 0, stream>>>(
      Wpos, wpos_b, (long)HDIM * EE);
  cvt_f32_bf16<<<(unsigned)(((long)EE * HDIM + 255) / 256), 256, 0, stream>>>(
      Wout, wout_b, (long)EE * HDIM);
  pe_build<<<(unsigned)(((long)TT * EE + 255) / 256), 256, 0, stream>>>(pe_b);

  // heads = we @ Wqkv^T   [8192 x 3072]
  gemm_nt_kernel<<<dim3(M1 / 128, 3 * HDIM / 64), 256, 0, stream>>>(
      we_b, EE, wqkv_b, EE, EE, heads_b, 3 * HDIM, nullptr, 0);
  // Rk = pe @ Wpos^T      [2048 x 1024]
  gemm_nt_kernel<<<dim3(TT / 128, HDIM / 64), 256, 0, stream>>>(
      pe_b, EE, wpos_b, EE, EE, rk_b, HDIM, nullptr, 0);

  prep_kernel<<<(unsigned)(((long)BB * HH * TT * DD + 255) / 256), 256, 0, stream>>>(
      heads_b, rk_b, r_w, r_r, qw_b, qr_b, kk_b, vt_b, rkh_b);

  attn_kernel<<<dim3(TT / 128, BB * HH), 256, 0, stream>>>(qw_b, qr_b, kk_b, vt_b,
                                                           rkh_b, lens, vec_b);

  // out = vec @ Wout^T, fused transpose to [B,T,E] f32
  gemm_nt_kernel<<<dim3(M1 / 128, EE / 64), 256, 0, stream>>>(
      vec_b, HDIM, wout_b, HDIM, HDIM, nullptr, 0, out, 1);
}
